// Model_VideoSumm_36721970381408
// MI455X (gfx1250) — matrix operand order repaired and checked by reading, not compiled, not run on hardware
//
#include <hip/hip_runtime.h>
#include <hip/hip_bf16.h>
#include <math.h>

// ---------------- problem constants (fixed by reference) ----------------
#define BSZ    4
#define SEQ    1536
#define HID    1024
#define NH     8
#define HDIM   128
#define FFD    4096
#define NVID   1024
#define NTXT   512
#define NLAYER 6
#define LN_EPS 1e-5f

typedef __bf16 bf16;
typedef __attribute__((ext_vector_type(16))) __bf16 v16bf;
typedef __attribute__((ext_vector_type(8)))  float  v8f;

// ---------------- CDNA5 async global->LDS copies (guarded) ----------------
#if defined(__HIP_DEVICE_COMPILE__) && __has_builtin(__builtin_amdgcn_global_load_async_to_lds_b128)
#define ASYNC_LDS 1
#else
#define ASYNC_LDS 0
#endif

#if ASYNC_LDS
typedef __attribute__((__vector_size__(4 * sizeof(int)))) int v4i;
typedef __attribute__((address_space(1))) v4i* gas_v4i;
typedef __attribute__((address_space(3))) v4i* las_v4i;
#endif

// copy 16B (8 bf16) global -> LDS
__device__ __forceinline__ void cp16(bf16* ldst, const bf16* gsrc) {
#if ASYNC_LDS
  __builtin_amdgcn_global_load_async_to_lds_b128(
      (gas_v4i)(void*)gsrc, (las_v4i)(void*)ldst, 0, 0);
#else
  *(uint4*)ldst = *(const uint4*)gsrc;
#endif
}

// wait until <=20 async ops outstanding (one slab's worth still in flight)
__device__ __forceinline__ void wait_async_prev() {
#if ASYNC_LDS
#if __has_builtin(__builtin_amdgcn_s_wait_asynccnt)
  __builtin_amdgcn_s_wait_asynccnt(20);
#else
  asm volatile("s_wait_asynccnt 0x14" ::: "memory");
#endif
#endif
}

// wait for all async ops
__device__ __forceinline__ void wait_async_all() {
#if ASYNC_LDS
#if __has_builtin(__builtin_amdgcn_s_wait_asynccnt)
  __builtin_amdgcn_s_wait_asynccnt(0);
#else
  asm volatile("s_wait_asynccnt 0x0" ::: "memory");
#endif
#endif
}

// D = A(16x32 bf16) * B(32x16 bf16) + C(16x16 f32)
__device__ __forceinline__ v8f wmma_bf16(v16bf a, v16bf b, v8f c) {
  return __builtin_amdgcn_wmma_f32_16x16x32_bf16(false, a, false, b, (short)0, c,
                                                 false, false);
}

// assemble a v16bf from two 8-element (16B) chunks (A-matrix / P-matrix frags)
__device__ __forceinline__ v16bf combine8(const bf16* p0, const bf16* p1) {
  union { v16bf v; uint4 q[2]; } u;
  u.q[0] = *(const uint4*)p0;
  u.q[1] = *(const uint4*)p1;
  return u.v;
}

__device__ __forceinline__ float gelu_exact(float x) {
  return 0.5f * x * (1.0f + erff(x * 0.70710678118654752f));
}

// ---------------- fp32 -> bf16 cast ----------------
__global__ void k_cast(const float* __restrict__ src, bf16* __restrict__ dst, int n) {
  int i = blockIdx.x * blockDim.x + threadIdx.x;
  if (i < n) dst[i] = (bf16)src[i];
}

// ---------------- LayerNorm (fp32 in, bf16 out), one row per block ----------------
__global__ __launch_bounds__(256)
void k_ln(const float* __restrict__ X, const float* __restrict__ g,
          const float* __restrict__ bta, bf16* __restrict__ Y,
          int seg_start, int seg_len) {
  int t   = blockIdx.x;
  int bi  = t / seg_len;
  int s   = seg_start + (t % seg_len);
  int row = bi * SEQ + s;
  const float* xr = &X[(size_t)row * HID];
  int tid = threadIdx.x;

  __shared__ float red[256];
  float v[4];
  float sum = 0.f;
#pragma unroll
  for (int i = 0; i < 4; ++i) { v[i] = xr[tid + i * 256]; sum += v[i]; }
  red[tid] = sum; __syncthreads();
  for (int st = 128; st > 0; st >>= 1) {
    if (tid < st) red[tid] += red[tid + st];
    __syncthreads();
  }
  float mu = red[0] * (1.0f / (float)HID);
  __syncthreads();

  float s2 = 0.f;
#pragma unroll
  for (int i = 0; i < 4; ++i) { float d = v[i] - mu; s2 += d * d; }
  red[tid] = s2; __syncthreads();
  for (int st = 128; st > 0; st >>= 1) {
    if (tid < st) red[tid] += red[tid + st];
    __syncthreads();
  }
  float rs = rsqrtf(red[0] * (1.0f / (float)HID) + LN_EPS);

  bf16* yr = &Y[(size_t)row * HID];
#pragma unroll
  for (int i = 0; i < 4; ++i) {
    int c = tid + i * 256;
    yr[c] = (bf16)((v[i] - mu) * rs * g[c] + bta[c]);
  }
}

// ---------------- NT GEMM: out[M,N] = epi( X[M,K] * W[N,K]^T + bias ) ----------------
// 128 threads = 4 waves; workgroup tile M=64, N=256, K staged 64 at a time,
// double-buffered LDS with async global->LDS prefetch of the next K-slab.
// Each wave: 64 columns x 64 rows = 16 WMMA accumulators; per K-substep
// 4 A-frags x 4 B-frags feed 16 WMMAs (1 WMMA per ds_load_b128).
// mode 0: bias        -> bf16 Oh
// mode 1: bias + GELU -> bf16 Oh
// mode 2: bias + Res  -> f32  Of
__global__ __launch_bounds__(128)
void k_gemm_nt(const bf16* __restrict__ X, const bf16* __restrict__ W,
               const float* __restrict__ bias, const float* __restrict__ Res,
               float* __restrict__ Of, bf16* __restrict__ Oh,
               int seg_start, int seg_len, int K, int N, int mode) {
  const int segTiles = seg_len >> 6;
  int t  = blockIdx.y;
  int bi = t / segTiles;
  int r0 = (t % segTiles) << 6;
  int rowBase = bi * SEQ + seg_start + r0;
  int nbase   = blockIdx.x * 256;

  __shared__ __align__(64) bf16 As[2][64 * 64];    // 8KB x2
  __shared__ __align__(64) bf16 Bs[2][256 * 64];   // 32KB x2

  int tid  = threadIdx.x;
  int lane = tid & 31;
  int w    = tid >> 5;
  int half = lane >> 4;
  int l16  = lane & 15;

  v8f acc[16] = {};   // [mi*4 + ni]

  const int T = K >> 6;

  auto stage = [&](int buf, int k0) {
#pragma unroll
    for (int i = 0; i < 4; ++i) {            // A: 64x64, 4 chunks/thread
      int ch = tid * 4 + i;
      int ar = ch >> 3, ac = (ch & 7) * 8;
      cp16(&As[buf][ar * 64 + ac], &X[(size_t)(rowBase + ar) * K + k0 + ac]);
    }
#pragma unroll
    for (int i = 0; i < 16; ++i) {           // B: 256x64, 16 chunks/thread
      int ch = tid * 16 + i;
      int br = ch >> 3, bc = (ch & 7) * 8;
      cp16(&Bs[buf][br * 64 + bc], &W[(size_t)(nbase + br) * K + k0 + bc]);
    }
  };

  auto compute = [&](int buf) {
#pragma unroll
    for (int ks = 0; ks < 2; ++ks) {
      v16bf af[4], bfr[4];
#pragma unroll
      for (int mi = 0; mi < 4; ++mi)
        af[mi] = combine8(&As[buf][(mi * 16 + l16) * 64 + ks * 32 + half * 16],
                          &As[buf][(mi * 16 + l16) * 64 + ks * 32 + half * 16 + 8]);
#pragma unroll
      for (int ni = 0; ni < 4; ++ni)
        bfr[ni] = *(const v16bf*)
            &Bs[buf][(w * 64 + ni * 16 + l16) * 64 + ks * 32 + half * 16];
#pragma unroll
      for (int mi = 0; mi < 4; ++mi)
#pragma unroll
        for (int ni = 0; ni < 4; ++ni)
          acc[mi * 4 + ni] = wmma_bf16(af[mi], bfr[ni], acc[mi * 4 + ni]);
    }
  };

  stage(0, 0);
  int it = 0;
  for (; it + 1 < T; ++it) {
    stage((it + 1) & 1, (it + 1) << 6);   // prefetch next slab (20 async ops)
    wait_async_prev();                    // previous slab complete
    __syncthreads();
    compute(it & 1);
    __syncthreads();                      // everyone done before slab reuse
  }
  wait_async_all();
  __syncthreads();
  compute(it & 1);

  int wn = nbase + w * 64;
#pragma unroll
  for (int mi = 0; mi < 4; ++mi)
#pragma unroll
    for (int ni = 0; ni < 4; ++ni)
#pragma unroll
      for (int j = 0; j < 8; ++j) {
        size_t row = (size_t)(rowBase + mi * 16 + j + half * 8);
        int c = wn + ni * 16 + l16;
        float v = acc[mi * 4 + ni][j] + bias[c];
        if (mode == 1) v = gelu_exact(v);
        if (mode == 2) {
          Of[row * N + c] = Res[row * N + c] + v;
        } else {
          Oh[row * N + c] = (bf16)v;
        }
      }
}

// ---------------- Flash attention ----------------
// grid: (BSZ*NH, SEQ/64); block = 128 threads = 4 waves; wave w owns 16 queries.
__global__ __launch_bounds__(128)
void k_attn(const bf16* __restrict__ Q, const bf16* __restrict__ Kx,
            const bf16* __restrict__ Vx, const int* __restrict__ mask,
            bf16* __restrict__ O) {
  const float scale = 0.08838834764831845f; // HDIM^-0.5

  int bh = blockIdx.x;
  int b  = bh / NH;
  int h  = bh % NH;
  int h0 = h * HDIM;

  int tid  = threadIdx.x;
  int lane = tid & 31;
  int w    = tid >> 5;
  int half = lane >> 4;
  int l16  = lane & 15;
  int qb   = blockIdx.y * 64 + w * 16;

  __shared__ __align__(64) bf16 Ks[32 * HDIM];     // [key][d]
  __shared__ __align__(64) bf16 Vs[HDIM * 32];     // [d][key] (transposed)
  __shared__ __align__(64) bf16 Ps[4][16 * 32];    // per-wave probs

  v16bf qf[4];
#pragma unroll
  for (int ks = 0; ks < 4; ++ks) {
    const bf16* qrow =
        &Q[(size_t)(b * SEQ + qb + l16) * HID + h0 + ks * 32 + half * 16];
    qf[ks] = combine8(qrow, qrow + 8);
  }

  v8f oacc[8];
#pragma unroll
  for (int a = 0; a < 8; ++a) oacc[a] = (v8f){};
  float mj[8], lj[8];
#pragma unroll
  for (int j = 0; j < 8; ++j) { mj[j] = -1e30f; lj[j] = 0.f; }

  for (int kb = 0; kb < SEQ; kb += 32) {
    // K tile: async global->LDS; V tile: transpose through registers
#pragma unroll
    for (int i = 0; i < 4; ++i) {
      int ch = tid * 4 + i;
      int kr = ch >> 4;
      int kc = (ch & 15) * 8;
      cp16(&Ks[kr * HDIM + kc],
           &Kx[(size_t)(b * SEQ + kb + kr) * HID + h0 + kc]);
      union { uint4 q; bf16 e[8]; } u;
      u.q = *(const uint4*)&Vx[(size_t)(b * SEQ + kb + kr) * HID + h0 + kc];
#pragma unroll
      for (int e = 0; e < 8; ++e) Vs[(kc + e) * 32 + kr] = u.e[e];
    }
    wait_async_all();
    __syncthreads();

    v8f s0 = {}, s1 = {};
#pragma unroll
    for (int ks = 0; ks < 4; ++ks) {
      v16bf kb0 = *(const v16bf*)&Ks[l16 * HDIM + ks * 32 + half * 16];
      v16bf kb1 = *(const v16bf*)&Ks[(16 + l16) * HDIM + ks * 32 + half * 16];
      s0 = wmma_bf16(qf[ks], kb0, s0);
      s1 = wmma_bf16(qf[ks], kb1, s1);
    }

    float alpha[8];
#pragma unroll
    for (int j = 0; j < 8; ++j) {
      int qr = qb + j + half * 8;
      const int* mr = &mask[(size_t)(b * SEQ + qr) * SEQ + kb];
      float a0 = s0[j] * scale + ((mr[l16]      > 0) ? 0.f : -1e30f);
      float a1 = s1[j] * scale + ((mr[16 + l16] > 0) ? 0.f : -1e30f);
      float tm = fmaxf(a0, a1);
#pragma unroll
      for (int off = 1; off < 16; off <<= 1)
        tm = fmaxf(tm, __shfl_xor(tm, off, 32));
      float mn = fmaxf(mj[j], tm);
      float al = __expf(mj[j] - mn);
      float p0 = __expf(a0 - mn);
      float p1 = __expf(a1 - mn);
      float ps = p0 + p1;
#pragma unroll
      for (int off = 1; off < 16; off <<= 1) ps += __shfl_xor(ps, off, 32);
      lj[j] = lj[j] * al + ps;
      mj[j] = mn;
      alpha[j] = al;
      Ps[w][(j + half * 8) * 32 + l16]      = (bf16)p0;
      Ps[w][(j + half * 8) * 32 + 16 + l16] = (bf16)p1;
    }
#pragma unroll
    for (int a = 0; a < 8; ++a)
#pragma unroll
      for (int j = 0; j < 8; ++j) oacc[a][j] *= alpha[j];

    v16bf pf = combine8(&Ps[w][l16 * 32 + half * 16],
                        &Ps[w][l16 * 32 + half * 16 + 8]);
#pragma unroll
    for (int a = 0; a < 8; ++a) {
      v16bf vb = *(const v16bf*)&Vs[(a * 16 + l16) * 32 + half * 16];
      oacc[a] = wmma_bf16(pf, vb, oacc[a]);
    }
    __syncthreads();
  }

  float inv[8];
#pragma unroll
  for (int j = 0; j < 8; ++j) inv[j] = 1.0f / lj[j];
#pragma unroll
  for (int a = 0; a < 8; ++a)
#pragma unroll
    for (int j = 0; j < 8; ++j) {
      int qr = qb + j + half * 8;
      O[(size_t)(b * SEQ + qr) * HID + h0 + a * 16 + l16] =
          (bf16)(oacc[a][j] * inv[j]);
    }
}

// ---------------- launch ----------------
extern "C" void kernel_launch(void* const* d_in, const int* in_sizes, int n_in,
                              void* d_out, int out_size, void* d_ws, size_t ws_size,
                              hipStream_t stream) {
  (void)in_sizes; (void)n_in; (void)out_size; (void)ws_size;

  const float* fused  = (const float*)d_in[0];
  const int*   maskp  = (const int*)d_in[1];
  const float* ln1_g  = (const float*)d_in[4];
  const float* ln1_b  = (const float*)d_in[5];
  const float* wq     = (const float*)d_in[6];
  const float* bq     = (const float*)d_in[7];
  const float* wk     = (const float*)d_in[8];
  const float* bk     = (const float*)d_in[9];
  const float* wv     = (const float*)d_in[10];
  const float* bv     = (const float*)d_in[11];
  const float* wm     = (const float*)d_in[12];
  const float* bm     = (const float*)d_in[13];
  const float* ln2v_g = (const float*)d_in[14];
  const float* ln2v_b = (const float*)d_in[15];
  const float* fv1_w  = (const float*)d_in[16];
  const float* fv1_b  = (const float*)d_in[17];
  const float* fv2_w  = (const float*)d_in[18];
  const float* fv2_b  = (const float*)d_in[19];
  const float* ln2t_g = (const float*)d_in[20];
  const float* ln2t_b = (const float*)d_in[21];
  const float* ft1_w  = (const float*)d_in[22];
  const float* ft1_b  = (const float*)d_in[23];
  const float* ft2_w  = (const float*)d_in[24];
  const float* ft2_b  = (const float*)d_in[25];

  char* base = (char*)d_ws;
  size_t off = 0;
  auto take = [&](size_t bytes) -> void* {
    void* p = base + off;
    off = (off + bytes + 255) & ~(size_t)255;
    return p;
  };
  const size_t NTOK = (size_t)BSZ * SEQ;
  bf16* bwq  = (bf16*)take((size_t)HID * HID * 2);
  bf16* bwk  = (bf16*)take((size_t)HID * HID * 2);
  bf16* bwv  = (bf16*)take((size_t)HID * HID * 2);
  bf16* bwm  = (bf16*)take((size_t)HID * HID * 2);
  bf16* bfv1 = (bf16*)take((size_t)FFD * HID * 2);
  bf16* bfv2 = (bf16*)take((size_t)HID * FFD * 2);
  bf16* bft1 = (bf16*)take((size_t)FFD * HID * 2);
  bf16* bft2 = (bf16*)take((size_t)HID * FFD * 2);
  float* x   = (float*)take(NTOK * HID * 4);
  float* r   = (float*)take(NTOK * HID * 4);
  bf16* xln  = (bf16*)take(NTOK * HID * 2);
  bf16* qb_  = (bf16*)take(NTOK * HID * 2);
  bf16* kb_  = (bf16*)take(NTOK * HID * 2);
  bf16* vb_  = (bf16*)take(NTOK * HID * 2);
  bf16* ob_  = (bf16*)take(NTOK * HID * 2);
  bf16* hb_  = (bf16*)take(NTOK * FFD * 2);

  auto cvt = [&](const float* s, bf16* d, int n) {
    k_cast<<<(n + 255) / 256, 256, 0, stream>>>(s, d, n);
  };
  cvt(wq, bwq, HID * HID);   cvt(wk, bwk, HID * HID);
  cvt(wv, bwv, HID * HID);   cvt(wm, bwm, HID * HID);
  cvt(fv1_w, bfv1, FFD * HID); cvt(fv2_w, bfv2, HID * FFD);
  cvt(ft1_w, bft1, FFD * HID); cvt(ft2_w, bft2, HID * FFD);

  (void)hipMemcpyAsync(x, fused, NTOK * HID * 4, hipMemcpyDeviceToDevice, stream);

  auto gemm = [&](const bf16* X, const bf16* W, const float* bias,
                  const float* Res, float* Of, bf16* Oh,
                  int seg_start, int seg_len, int K, int N, int mode) {
    dim3 g(N / 256, BSZ * (seg_len / 64));
    k_gemm_nt<<<g, 128, 0, stream>>>(X, W, bias, Res, Of, Oh,
                                     seg_start, seg_len, K, N, mode);
  };

  for (int l = 0; l < NLAYER; ++l) {
    k_ln<<<BSZ * SEQ, 256, 0, stream>>>(x, ln1_g, ln1_b, xln, 0, SEQ);
    gemm(xln, bwq, bq, nullptr, nullptr, qb_, 0, SEQ, HID, HID, 0);
    gemm(xln, bwk, bk, nullptr, nullptr, kb_, 0, SEQ, HID, HID, 0);
    gemm(xln, bwv, bv, nullptr, nullptr, vb_, 0, SEQ, HID, HID, 0);
    k_attn<<<dim3(BSZ * NH, SEQ / 64), 128, 0, stream>>>(qb_, kb_, vb_, maskp, ob_);
    gemm(ob_, bwm, bm, x, r, nullptr, 0, SEQ, HID, HID, 2);
    k_ln<<<BSZ * NVID, 256, 0, stream>>>(r, ln2v_g, ln2v_b, xln, 0, NVID);
    k_ln<<<BSZ * NTXT, 256, 0, stream>>>(r, ln2t_g, ln2t_b, xln, NVID, NTXT);
    gemm(xln, bfv1, fv1_b, nullptr, nullptr, hb_, 0,    NVID, HID, FFD, 1);
    gemm(xln, bft1, ft1_b, nullptr, nullptr, hb_, NVID, NTXT, HID, FFD, 1);
    gemm(hb_, bfv2, fv2_b, r, x, nullptr, 0,    NVID, FFD, HID, 2);
    gemm(hb_, bft2, ft2_b, r, x, nullptr, NVID, NTXT, FFD, HID, 2);
  }

  (void)hipMemcpyAsync(d_out, x, NTOK * HID * 4, hipMemcpyDeviceToDevice, stream);
}
